// LayoutGraphModel_74191265071128
// MI455X (gfx1250) — matrix operand order repaired and checked by reading, not compiled, not run on hardware
//
#include <hip/hip_runtime.h>
#include <hip/hip_bf16.h>

#define NN 32768
#define EE 262144
#define SS 128

typedef __attribute__((ext_vector_type(16))) __bf16 v16bf;
typedef __attribute__((ext_vector_type(8)))  float  v8f;
typedef __attribute__((ext_vector_type(4)))  unsigned int u32x4;
typedef __attribute__((ext_vector_type(4)))  int v4i;

union Frag { u32x4 u[2]; v16bf v; };

__device__ __forceinline__ unsigned short f2bf(float f) {
    unsigned int u = __float_as_uint(f);
    u += 0x7FFFu + ((u >> 16) & 1u);
    return (unsigned short)(u >> 16);
}
__device__ __forceinline__ float bf2f(unsigned short h) {
    return __uint_as_float(((unsigned int)h) << 16);
}

// ---- CDNA5 async global->LDS path (guarded; falls back to sync staging) ----
#if __has_builtin(__builtin_amdgcn_global_load_async_to_lds_b128)
#define HAVE_ASYNC_LDS 1
#else
#define HAVE_ASYNC_LDS 0
#endif

#if __has_builtin(__builtin_amdgcn_s_wait_asynccnt)
#define ASYNC_WAIT(n) __builtin_amdgcn_s_wait_asynccnt(n)
#else
#define ASYNC_WAIT(n) asm volatile("s_wait_asynccnt %0" ::"i"(n) : "memory")
#endif

typedef __attribute__((address_space(1))) v4i* gv4i_p;   // global v4i*
typedef __attribute__((address_space(3))) v4i* lv4i_p;   // LDS v4i*
#define GAS(p) ((gv4i_p)(p))
#define LAS(p) ((lv4i_p)(p))

// ---------------- degree / CSR build ----------------
__global__ void deg_count_k(const int* __restrict__ dst, int* __restrict__ degI) {
    int e = blockIdx.x * 256 + threadIdx.x;
    if (e < EE) atomicAdd(&degI[dst[e]], 1);
}
__global__ void inv_deg_k(const int* __restrict__ degI, float* __restrict__ invd) {
    int i = blockIdx.x * 256 + threadIdx.x;
    if (i < NN) invd[i] = 1.0f / fmaxf((float)degI[i], 1.0f);
}
__global__ void scan_block_sums_k(const int* __restrict__ degI, int* __restrict__ bsum) {
    __shared__ int sh[256];
    int t = threadIdx.x;
    sh[t] = degI[blockIdx.x * 256 + t];
    __syncthreads();
    for (int o = 128; o > 0; o >>= 1) { if (t < o) sh[t] += sh[t + o]; __syncthreads(); }
    if (t == 0) bsum[blockIdx.x] = sh[0];
}
__global__ void scan_top_k(int* __restrict__ bsum) {  // 1 block, 128 threads, -> exclusive
    __shared__ int sh[128];
    int t = threadIdx.x;
    int orig = bsum[t];
    sh[t] = orig; __syncthreads();
    for (int o = 1; o < 128; o <<= 1) {
        int v = (t >= o) ? sh[t - o] : 0;
        __syncthreads(); sh[t] += v; __syncthreads();
    }
    bsum[t] = sh[t] - orig;
}
__global__ void scan_offsets_k(const int* __restrict__ degI, const int* __restrict__ boff,
                               int* __restrict__ offs, int* __restrict__ cursor) {
    __shared__ int sh[256];
    int t = threadIdx.x, i = blockIdx.x * 256 + t;
    int d = degI[i];
    sh[t] = d; __syncthreads();
    for (int o = 1; o < 256; o <<= 1) {
        int v = (t >= o) ? sh[t - o] : 0;
        __syncthreads(); sh[t] += v; __syncthreads();
    }
    int excl = boff[blockIdx.x] + sh[t] - d;
    offs[i] = excl; cursor[i] = excl;
}
__global__ void csr_fill_k(const int* __restrict__ src, const int* __restrict__ dst,
                           int* __restrict__ cursor, int* __restrict__ csrSrc) {
    int e = blockIdx.x * 256 + threadIdx.x;
    if (e < EE) { int p = atomicAdd(&cursor[dst[e]], 1); csrSrc[p] = src[e]; }
}

// ---------------- feature concat -> bf16 (N x 192, 190 valid) ----------------
__global__ void concat_x0_k(const float* __restrict__ nf, const float* __restrict__ ncf,
                            const float* __restrict__ emb, const int* __restrict__ ops,
                            unsigned short* __restrict__ X0) {
    int idx = blockIdx.x * 256 + threadIdx.x;
    if (idx >= NN * 192) return;
    int n = idx / 192, c = idx - n * 192;
    float v;
    if (c < 90)       v = nf[n * 90 + c];
    else if (c < 126) v = ncf[n * 36 + (c - 90)];
    else if (c < 190) v = emb[ops[n] * 64 + (c - 126)];
    else              v = 0.f;
    X0[idx] = f2bf(v);
}

// ---------------- weight fp32 -> bf16 (padded) ----------------
__global__ void conv_w_k(const float* __restrict__ W, int Dout, int Din, int DinP,
                         unsigned short* __restrict__ Wb, int total) {
    int idx = blockIdx.x * 256 + threadIdx.x;
    if (idx >= total) return;
    int j = idx / DinP, k = idx - j * DinP;
    float v = (j < Dout && k < Din) ? W[j * Din + k] : 0.f;
    Wb[idx] = f2bf(v);
}
__global__ void pad_bias_k(const float* __restrict__ b, int D, int DP, float* __restrict__ bp) {
    int i = blockIdx.x * 256 + threadIdx.x;
    if (i < DP) bp[i] = (i < D) ? b[i] : 0.f;
}

// ---------------- WMMA bf16 GEMM: C[M,Nout] = act(A@W^T [+ A2@W2^T] + bias) ----------------
// 128x128 block tile, 8 waves of 32, each wave = 32x64 patch = 2x4 wmma tiles, K step 32.
// Double-buffered LDS tiles filled by CDNA5 async global->LDS copies (ASYNCcnt).
#define LSTR 40  // LDS row stride in bf16 elems (80B) -> conflict-free b128 reads

__device__ __forceinline__ void stage_tile(const unsigned short* __restrict__ Ap,
                                           const unsigned short* __restrict__ Wp,
                                           int ld, int k0, int m0, int n0, int t,
                                           unsigned short* AsB, unsigned short* BsB) {
#pragma unroll
    for (int i = 0; i < 2; ++i) {
        int idx = t + i * 256;          // 0..511
        int row = idx >> 2;             // 0..127
        int c   = (idx & 3) * 8;        // bf16 chunk offset (16B)
#if HAVE_ASYNC_LDS
        __builtin_amdgcn_global_load_async_to_lds_b128(
            GAS(Ap + (size_t)(m0 + row) * ld + k0 + c), LAS(&AsB[row * LSTR + c]), 0, 0);
        __builtin_amdgcn_global_load_async_to_lds_b128(
            GAS(Wp + (size_t)(n0 + row) * ld + k0 + c), LAS(&BsB[row * LSTR + c]), 0, 0);
#else
        *(u32x4*)(&AsB[row * LSTR + c]) = *(const u32x4*)(Ap + (size_t)(m0 + row) * ld + k0 + c);
        *(u32x4*)(&BsB[row * LSTR + c]) = *(const u32x4*)(Wp + (size_t)(n0 + row) * ld + k0 + c);
#endif
    }
}

__global__ __launch_bounds__(256) void gemm_bf16_wmma_k(
    const unsigned short* __restrict__ A,  const unsigned short* __restrict__ W,
    const unsigned short* __restrict__ A2, const unsigned short* __restrict__ W2,
    const float* __restrict__ bias, float* __restrict__ C,
    int Nout, int K1, int K2, int actLeaky)
{
    __shared__ unsigned short As[2][128 * LSTR];
    __shared__ unsigned short Bs[2][128 * LSTR];

    const int t    = threadIdx.x;
    const int lane = t & 31;
    const int wid  = t >> 5;
    const int g    = lane >> 4;    // half-wave group
    const int ln   = lane & 15;
    const int wm   = (wid >> 1) * 32;
    const int wn   = (wid & 1) * 64;
    const int m0   = blockIdx.y * 128;
    const int n0   = blockIdx.x * 128;

    v8f acc[2][4];
    v8f vz = {0.f,0.f,0.f,0.f,0.f,0.f,0.f,0.f};
    #pragma unroll
    for (int i = 0; i < 2; ++i)
        #pragma unroll
        for (int j = 0; j < 4; ++j) acc[i][j] = vz;

    const int steps1 = K1 >> 5;
    const int steps2 = K2 >> 5;
    const int steps  = steps1 + steps2;

    // prologue: kick off stage 0 into buffer 0
    {
        const unsigned short* Ap = (0 < steps1) ? A : A2;
        const unsigned short* Wp = (0 < steps1) ? W : W2;
        int ld = (0 < steps1) ? K1 : K2;
        stage_tile(Ap, Wp, ld, 0, m0, n0, t, As[0], Bs[0]);
    }

    for (int s = 0; s < steps; ++s) {
        const int cur = s & 1;
        if (s + 1 < steps) {
            // issue next tile into the other buffer, then wait for current tile only
            const unsigned short* Ap; const unsigned short* Wp; int k0, ld;
            int sn = s + 1;
            if (sn < steps1) { Ap = A;  Wp = W;  k0 = sn << 5;            ld = K1; }
            else             { Ap = A2; Wp = W2; k0 = (sn - steps1) << 5; ld = K2; }
            stage_tile(Ap, Wp, ld, k0, m0, n0, t, As[cur ^ 1], Bs[cur ^ 1]);
#if HAVE_ASYNC_LDS
            ASYNC_WAIT(4);
#endif
        } else {
#if HAVE_ASYNC_LDS
            ASYNC_WAIT(0);
#endif
        }
        __syncthreads();

        Frag af[2], bfr[4];
        #pragma unroll
        for (int am = 0; am < 2; ++am) {
            const unsigned short* base = &As[cur][(wm + am * 16 + ln) * LSTR];
            af[am].u[0] = *(const u32x4*)(base + g * 8);        // K = 8g..8g+7
            af[am].u[1] = *(const u32x4*)(base + 16 + g * 8);   // K = 16+8g..
        }
        #pragma unroll
        for (int bn = 0; bn < 4; ++bn) {
            const unsigned short* base = &Bs[cur][(wn + bn * 16 + ln) * LSTR];
            bfr[bn].u[0] = *(const u32x4*)(base + g * 8);       // K = 8g..8g+7
            bfr[bn].u[1] = *(const u32x4*)(base + 16 + g * 8);  // K = 16+8g..
        }
        #pragma unroll
        for (int am = 0; am < 2; ++am)
            #pragma unroll
            for (int bn = 0; bn < 4; ++bn)
                acc[am][bn] = __builtin_amdgcn_wmma_f32_16x16x32_bf16(
                    false, af[am].v, false, bfr[bn].v, (short)0, acc[am][bn], false, false);
        __syncthreads();
    }

    // epilogue: C layout lane ln = col, vgpr r -> row 8g+r
    #pragma unroll
    for (int am = 0; am < 2; ++am) {
        #pragma unroll
        for (int bn = 0; bn < 4; ++bn) {
            int n = n0 + wn + bn * 16 + ln;
            float bv = bias[n];
            #pragma unroll
            for (int r = 0; r < 8; ++r) {
                int m = m0 + wm + am * 16 + g * 8 + r;
                float v = acc[am][bn][r] + bv;
                if (actLeaky) v = (v >= 0.f) ? v : 0.01f * v;
                C[(size_t)m * Nout + n] = v;
            }
        }
    }
}

// ---------------- graphwise LayerNorm ----------------
__global__ void gln_stats_k(const float* __restrict__ H, int D, int DP,
                            float* __restrict__ segMean, float* __restrict__ segInv) {
    __shared__ float sh[512];
    int s = blockIdx.x, t = threadIdx.x;
    float sum = 0.f, ssq = 0.f;
    for (int r = 0; r < 256; ++r) {
        const float* row = H + (size_t)(s * 256 + r) * DP;
        for (int c = t; c < D; c += 256) { float v = row[c]; sum += v; ssq += v * v; }
    }
    sh[t] = sum; sh[256 + t] = ssq; __syncthreads();
    for (int o = 128; o > 0; o >>= 1) {
        if (t < o) { sh[t] += sh[t + o]; sh[256 + t] += sh[256 + t + o]; }
        __syncthreads();
    }
    if (t == 0) {
        float cnt = 256.0f * (float)D;
        float m = sh[0] / cnt;
        float var = sh[256] / cnt - m * m;
        segMean[s] = m; segInv[s] = rsqrtf(var + 1e-5f);
    }
}
__global__ void gln_apply_k(const float* __restrict__ H, int D, int DP,
                            const float* __restrict__ mean, const float* __restrict__ inv,
                            const float* __restrict__ gamma, const float* __restrict__ beta,
                            unsigned short* __restrict__ X) {
    int idx = blockIdx.x * 256 + threadIdx.x;
    if (idx >= NN * DP) return;
    int n = idx / DP, c = idx - n * DP;
    float o = 0.f;
    if (c < D) {
        int s = n >> 8;
        o = (H[(size_t)n * DP + c] - mean[s]) * inv[s] * gamma[c] + beta[c];
    }
    X[idx] = f2bf(o);
}

// ---------------- SAGE mean-aggregate (CSR gather), D in {256,512} ----------------
__global__ void sage_agg_k(const unsigned short* __restrict__ Xb, int D,
                           const int* __restrict__ offs, const int* __restrict__ degI,
                           const float* __restrict__ invd, const int* __restrict__ csrSrc,
                           unsigned short* __restrict__ aggB) {
    int n = blockIdx.x, t = threadIdx.x;          // 128 threads
    int s0 = offs[n], cnt = degI[n];
    float w = invd[n];
    int nc = D >> 7;                               // 2 or 4
    float acc[4] = {0.f, 0.f, 0.f, 0.f};
    for (int i = 0; i < cnt; ++i) {
        const unsigned short* row = Xb + (size_t)csrSrc[s0 + i] * D;
        for (int j = 0; j < nc; ++j) acc[j] += bf2f(row[t + j * 128]);
    }
    unsigned short* out = aggB + (size_t)n * D;
    for (int j = 0; j < nc; ++j) out[t + j * 128] = f2bf(acc[j] * w);
}

// ---------------- PairNorm (D = 512 fixed), deterministic two-stage ----------------
__global__ void pn_colsum_part_k(const float* __restrict__ H, float* __restrict__ colpart) {
    int t = threadIdx.x;
    float a0 = 0.f, a1 = 0.f;
    size_t r0 = (size_t)blockIdx.x * 128;
    for (int r = 0; r < 128; ++r) {
        const float* row = H + (r0 + r) * 512;
        a0 += row[t]; a1 += row[t + 256];
    }
    colpart[(size_t)blockIdx.x * 512 + t] = a0;
    colpart[(size_t)blockIdx.x * 512 + t + 256] = a1;
}
__global__ void pn_colsum_final_k(const float* __restrict__ colpart, float* __restrict__ colsum) {
    int c = blockIdx.x * 256 + threadIdx.x;
    if (c >= 512) return;
    float s = 0.f;
    for (int b = 0; b < 256; ++b) s += colpart[(size_t)b * 512 + c];
    colsum[c] = s;
}
__global__ void pn_norm2_part_k(const float* __restrict__ H, const float* __restrict__ colsum,
                                float* __restrict__ part) {
    __shared__ float sh[256];
    float local = 0.f;
    const float invN = 1.0f / (float)NN;
    for (size_t i = (size_t)blockIdx.x * 256 + threadIdx.x; i < (size_t)NN * 512;
         i += (size_t)gridDim.x * 256) {
        int c = (int)(i & 511);
        float v = H[i] - colsum[c] * invN;
        local += v * v;
    }
    sh[threadIdx.x] = local; __syncthreads();
    for (int o = 128; o > 0; o >>= 1) { if (threadIdx.x < o) sh[threadIdx.x] += sh[threadIdx.x + o]; __syncthreads(); }
    if (threadIdx.x == 0) part[blockIdx.x] = sh[0];
}
__global__ void reduce512_k(const float* __restrict__ part, float* __restrict__ out) {
    __shared__ float sh[512];
    int t = threadIdx.x;
    sh[t] = part[t]; __syncthreads();
    for (int o = 256; o > 0; o >>= 1) { if (t < o) sh[t] += sh[t + o]; __syncthreads(); }
    if (t == 0) out[0] = sh[0];
}
__global__ void pn_apply_k(const float* __restrict__ H, const float* __restrict__ colsum,
                           const float* __restrict__ norm2, int relu,
                           unsigned short* __restrict__ X) {
    size_t idx = (size_t)blockIdx.x * 256 + threadIdx.x;
    if (idx >= (size_t)NN * 512) return;
    int c = (int)(idx & 511);
    const float invN = 1.0f / (float)NN;
    float scale = 1.0f / (1e-5f + sqrtf(norm2[0] * invN));
    float v = (H[idx] - colsum[c] * invN) * scale;
    if (relu) v = fmaxf(v, 0.f);
    X[idx] = f2bf(v);
}

// ---------------- pooling + head ----------------
__global__ void segpool_k(const float* __restrict__ H, float* __restrict__ segSums) {
    int s = blockIdx.x, t = threadIdx.x;
    float a0 = 0.f, a1 = 0.f;
    size_t base = (size_t)s * 256;
    for (int r = 0; r < 256; ++r) {
        const float* row = H + (base + r) * 512;
        a0 += row[t]; a1 += row[t + 256];
    }
    segSums[(size_t)s * 512 + t] = a0;
    segSums[(size_t)s * 512 + t + 256] = a1;
}
__global__ void head_k(const float* __restrict__ segSums, const float* __restrict__ wc,
                       const float* __restrict__ bc, float* __restrict__ out) {
    __shared__ float sh[256];
    int s = blockIdx.x, t = threadIdx.x;
    const float* row = segSums + (size_t)s * 512;
    sh[t] = row[t] * wc[t] + row[t + 256] * wc[t + 256];
    __syncthreads();
    for (int o = 128; o > 0; o >>= 1) { if (t < o) sh[t] += sh[t + o]; __syncthreads(); }
    if (t == 0) out[s] = sh[0] + bc[0];
}

extern "C" void kernel_launch(void* const* d_in, const int* in_sizes, int n_in,
                              void* d_out, int out_size, void* d_ws, size_t ws_size,
                              hipStream_t stream) {
    (void)in_sizes; (void)n_in; (void)out_size; (void)ws_size;
    const float* nf   = (const float*)d_in[0];
    const float* ncf  = (const float*)d_in[1];
    const float* emb  = (const float*)d_in[2];
    const float* w1   = (const float*)d_in[3];
    const float* b1   = (const float*)d_in[4];
    const float* g1   = (const float*)d_in[5];
    const float* be1  = (const float*)d_in[6];
    const float* w2   = (const float*)d_in[7];
    const float* b2   = (const float*)d_in[8];
    const float* g2   = (const float*)d_in[9];
    const float* be2  = (const float*)d_in[10];
    const float* wl1  = (const float*)d_in[11];
    const float* bl1  = (const float*)d_in[12];
    const float* wr1  = (const float*)d_in[13];
    const float* wl2  = (const float*)d_in[14];
    const float* bl2  = (const float*)d_in[15];
    const float* wr2  = (const float*)d_in[16];
    const float* wl3  = (const float*)d_in[17];
    const float* bl3  = (const float*)d_in[18];
    const float* wr3  = (const float*)d_in[19];
    const float* wc   = (const float*)d_in[20];
    const float* bc   = (const float*)d_in[21];
    const int* ops    = (const int*)d_in[22];
    const int* edges  = (const int*)d_in[23];
    const int* esrc   = edges;
    const int* edst   = edges + EE;
    float* out        = (float*)d_out;

    char* ws = (char*)d_ws;
    size_t off = 0;
    auto take = [&](size_t bytes) -> void* {
        void* p = ws + off;
        off += (bytes + 255) & ~(size_t)255;
        return p;
    };
    unsigned short* Xa    = (unsigned short*)take((size_t)NN * 512 * 2);
    unsigned short* AGGb  = (unsigned short*)take((size_t)NN * 512 * 2);
    float*          Hf    = (float*)take((size_t)NN * 512 * 4);
    unsigned short* W1b   = (unsigned short*)take(384 * 192 * 2);
    unsigned short* W2b   = (unsigned short*)take(256 * 384 * 2);
    unsigned short* Wl1b  = (unsigned short*)take(512 * 256 * 2);
    unsigned short* Wr1b  = (unsigned short*)take(512 * 256 * 2);
    unsigned short* Wl2b  = (unsigned short*)take(512 * 512 * 2);
    unsigned short* Wr2b  = (unsigned short*)take(512 * 512 * 2);
    unsigned short* Wl3b  = (unsigned short*)take(512 * 512 * 2);
    unsigned short* Wr3b  = (unsigned short*)take(512 * 512 * 2);
    float*          b1P   = (float*)take(384 * 4);
    int*            degI  = (int*)take((size_t)NN * 4);
    float*          invd  = (float*)take((size_t)NN * 4);
    int*            offs  = (int*)take((size_t)NN * 4);
    int*            curs  = (int*)take((size_t)NN * 4);
    int*            csrS  = (int*)take((size_t)EE * 4);
    int*            bsum  = (int*)take(128 * 4);
    float*          segM  = (float*)take(SS * 4);
    float*          segI  = (float*)take(SS * 4);
    float*          colpart = (float*)take((size_t)256 * 512 * 4);
    float*          colsum  = (float*)take(512 * 4);
    float*          pnpart  = (float*)take(512 * 4);
    float*          norm2   = (float*)take(256);
    float*          segSums = (float*)take((size_t)SS * 512 * 4);

    // ---- degree + CSR (once per call, reused by 3 SAGE layers) ----
    (void)hipMemsetAsync(degI, 0, (size_t)NN * 4, stream);
    deg_count_k<<<EE / 256, 256, 0, stream>>>(edst, degI);
    inv_deg_k<<<NN / 256, 256, 0, stream>>>(degI, invd);
    scan_block_sums_k<<<128, 256, 0, stream>>>(degI, bsum);
    scan_top_k<<<1, 128, 0, stream>>>(bsum);
    scan_offsets_k<<<128, 256, 0, stream>>>(degI, bsum, offs, curs);
    csr_fill_k<<<EE / 256, 256, 0, stream>>>(esrc, edst, curs, csrS);

    // ---- weights -> bf16 (padded) ----
    conv_w_k<<<(384 * 192 + 255) / 256, 256, 0, stream>>>(w1, 380, 190, 192, W1b, 384 * 192);
    conv_w_k<<<(256 * 384 + 255) / 256, 256, 0, stream>>>(w2, 256, 380, 384, W2b, 256 * 384);
    conv_w_k<<<(512 * 256 + 255) / 256, 256, 0, stream>>>(wl1, 512, 256, 256, Wl1b, 512 * 256);
    conv_w_k<<<(512 * 256 + 255) / 256, 256, 0, stream>>>(wr1, 512, 256, 256, Wr1b, 512 * 256);
    conv_w_k<<<(512 * 512 + 255) / 256, 256, 0, stream>>>(wl2, 512, 512, 512, Wl2b, 512 * 512);
    conv_w_k<<<(512 * 512 + 255) / 256, 256, 0, stream>>>(wr2, 512, 512, 512, Wr2b, 512 * 512);
    conv_w_k<<<(512 * 512 + 255) / 256, 256, 0, stream>>>(wl3, 512, 512, 512, Wl3b, 512 * 512);
    conv_w_k<<<(512 * 512 + 255) / 256, 256, 0, stream>>>(wr3, 512, 512, 512, Wr3b, 512 * 512);
    pad_bias_k<<<2, 256, 0, stream>>>(b1, 380, 384, b1P);

    // ---- concat features -> bf16 X (N x 192) ----
    concat_x0_k<<<NN * 192 / 256, 256, 0, stream>>>(nf, ncf, emb, ops, Xa);

    dim3 blk(256);
    // ---- MLP1: (N x 192) @ (384 x 192)^T, leaky ----
    gemm_bf16_wmma_k<<<dim3(384 / 128, NN / 128), blk, 0, stream>>>(
        Xa, W1b, nullptr, nullptr, b1P, Hf, 384, 192, 0, 1);
    gln_stats_k<<<SS, 256, 0, stream>>>(Hf, 380, 384, segM, segI);
    gln_apply_k<<<NN * 384 / 256, 256, 0, stream>>>(Hf, 380, 384, segM, segI, g1, be1, Xa);

    // ---- MLP2: (N x 384) @ (256 x 384)^T, leaky ----
    gemm_bf16_wmma_k<<<dim3(256 / 128, NN / 128), blk, 0, stream>>>(
        Xa, W2b, nullptr, nullptr, b2, Hf, 256, 384, 0, 1);
    gln_stats_k<<<SS, 256, 0, stream>>>(Hf, 256, 256, segM, segI);
    gln_apply_k<<<NN * 256 / 256, 256, 0, stream>>>(Hf, 256, 256, segM, segI, g2, be2, Xa);

    // ---- SAGE layer 1 (256 -> 512) ----
    sage_agg_k<<<NN, 128, 0, stream>>>(Xa, 256, offs, degI, invd, csrS, AGGb);
    gemm_bf16_wmma_k<<<dim3(512 / 128, NN / 128), blk, 0, stream>>>(
        AGGb, Wl1b, Xa, Wr1b, bl1, Hf, 512, 256, 256, 0);
    pn_colsum_part_k<<<256, 256, 0, stream>>>(Hf, colpart);
    pn_colsum_final_k<<<2, 256, 0, stream>>>(colpart, colsum);
    pn_norm2_part_k<<<512, 256, 0, stream>>>(Hf, colsum, pnpart);
    reduce512_k<<<1, 512, 0, stream>>>(pnpart, norm2);
    pn_apply_k<<<NN * 512 / 256, 256, 0, stream>>>(Hf, colsum, norm2, 1, Xa);

    // ---- SAGE layer 2 (512 -> 512) ----
    sage_agg_k<<<NN, 128, 0, stream>>>(Xa, 512, offs, degI, invd, csrS, AGGb);
    gemm_bf16_wmma_k<<<dim3(512 / 128, NN / 128), blk, 0, stream>>>(
        AGGb, Wl2b, Xa, Wr2b, bl2, Hf, 512, 512, 512, 0);
    pn_colsum_part_k<<<256, 256, 0, stream>>>(Hf, colpart);
    pn_colsum_final_k<<<2, 256, 0, stream>>>(colpart, colsum);
    pn_norm2_part_k<<<512, 256, 0, stream>>>(Hf, colsum, pnpart);
    reduce512_k<<<1, 512, 0, stream>>>(pnpart, norm2);
    pn_apply_k<<<NN * 512 / 256, 256, 0, stream>>>(Hf, colsum, norm2, 1, Xa);

    // ---- SAGE layer 3 (512 -> 512), no norm/act ----
    sage_agg_k<<<NN, 128, 0, stream>>>(Xa, 512, offs, degI, invd, csrS, AGGb);
    gemm_bf16_wmma_k<<<dim3(512 / 128, NN / 128), blk, 0, stream>>>(
        AGGb, Wl3b, Xa, Wr3b, bl3, Hf, 512, 512, 512, 0);

    // ---- segment-sum pooling + linear head ----
    segpool_k<<<SS, 256, 0, stream>>>(Hf, segSums);
    head_k<<<SS, 256, 0, stream>>>(segSums, wc, bc, out);
}
